// Hodge_SpatialConv_Pool_68702296866874
// MI455X (gfx1250) — compile-verified, matrix operand order checked by reading, not run on hardware
//
#include <hip/hip_runtime.h>
#include <hip/hip_bf16.h>

typedef __attribute__((ext_vector_type(16))) _Float16 v16h;
typedef __attribute__((ext_vector_type(8)))  float    v8f;

#define LEAKY_SLOPE 0.33f
#define BN_EPS 1e-5f

// CDNA5 f16 WMMA A/B fragment: K-index within a 32-wide k-block for element i
// of the v16h, given lane half (lane>>4).
// half 0: K = 0..7 then 16..23 ; half 1: K = 8..15 then 24..31
__device__ __forceinline__ int kidx16(int i, int half) {
  return (i & 7) + ((i >> 3) << 4) + (half << 3);
}

// ---------------------------------------------------------------------------
// Laguerre recurrence step, 1 channel:
// out[n] = (alpha*Tc[n] - sum_{j<16} w[n*16+j]*Tc[src[n*16+j]] - beta*Tp[n])*scale
// Exploits dst[e] == e/16 (fixed-degree, dst-sorted edge list) -> gather only.
// Edge data read as int4/float4 (64B per node, 4+4 loads instead of 16+16).
// ---------------------------------------------------------------------------
__global__ void lag_step_c1(const float* __restrict__ Tc, const float* __restrict__ Tp,
                            const int* __restrict__ src, const float* __restrict__ ew,
                            float* __restrict__ out, int Nn,
                            float alpha, float beta, float scale) {
  int n = blockIdx.x * blockDim.x + threadIdx.x;
  if (n >= Nn) return;
  const int4*   s4 = reinterpret_cast<const int4*>(src + n * 16);
  const float4* w4 = reinterpret_cast<const float4*>(ew + n * 16);
  float mv = 0.f;
#pragma unroll
  for (int j = 0; j < 4; ++j) {
    int4   s = s4[j];
    float4 w = w4[j];
    mv += w.x * Tc[s.x] + w.y * Tc[s.y] + w.z * Tc[s.z] + w.w * Tc[s.w];
  }
  out[n] = (alpha * Tc[n] - mv - beta * Tp[n]) * scale;
}

// Same, 32 channels: one wave per node, lane = channel.
// Edge (src,w) pairs are wave-uniform: lane j loads edge j once (one coalesced
// 64B transaction per wave), then values are distributed via __shfl (LDS
// crossbar) instead of 16 redundant broadcast loads per lane.
__global__ void lag_step_c32(const float* __restrict__ Tc, const float* __restrict__ Tp,
                             const int* __restrict__ src, const float* __restrict__ ew,
                             float* __restrict__ out, int Nn,
                             float alpha, float beta, float scale) {
  int t = blockIdx.x * blockDim.x + threadIdx.x;
  int n = t >> 5, c = t & 31;
  if (n >= Nn) return;
  int lane = threadIdx.x & 31;
  int base = n * 16;
  int   s_l = src[base + (lane & 15)];   // lanes 16-31 duplicate lanes 0-15
  float w_l = ew [base + (lane & 15)];
  float mv = 0.f;
#pragma unroll
  for (int j = 0; j < 16; ++j) {
    int   s = __shfl(s_l, j, 32);
    float w = __shfl(w_l, j, 32);
    mv += w * Tc[s * 32 + c];            // coalesced 128B gather row
  }
  out[t] = (alpha * Tc[t] - mv - beta * Tp[t]) * scale;
}

// Layer-0 dense mix (1 -> 32 channels): pure outer product, VALU is right here.
__global__ void lag_dense_in1(const float* __restrict__ x, const float* __restrict__ t1,
                              const float* __restrict__ t2, const float* __restrict__ t3,
                              const float* __restrict__ W, const float* __restrict__ bias,
                              float* __restrict__ out, int Nn) {
  int t = blockIdx.x * blockDim.x + threadIdx.x;
  int n = t >> 5, c = t & 31;
  if (n >= Nn) return;
  out[t] = x[n] * W[c] + t1[n] * W[32 + c] + t2[n] * W[64 + c] + t3[n] * W[96 + c] + bias[c];
}

// ---------------------------------------------------------------------------
// Layer-1 dense mix with WMMA: out[n,0:32] = sum_k T_k[n,:] @ W[k] + bias.
// W is (4,32,32) row-major == concatenated (128x32). One wave per 16x16 tile,
// 4 polys x K=32 each -> 4 v_wmma_f32_16x16x32_f16 per tile (f32 accumulate).
// ---------------------------------------------------------------------------
__global__ void lag_dense32_wmma(const float* __restrict__ T0, const float* __restrict__ T1,
                                 const float* __restrict__ T2, const float* __restrict__ T3,
                                 const float* __restrict__ W, const float* __restrict__ bias,
                                 float* __restrict__ out, int Nn) {
  int wave = (blockIdx.x * blockDim.x + threadIdx.x) >> 5;
  int tiles = (Nn >> 4) * 2;            // 2 column tiles of 16
  if (wave >= tiles) return;            // whole-wave exit: EXEC all-1 for WMMA
  int tm = wave >> 1, tn = wave & 1;
  const float* Ts[4] = {T0, T1, T2, T3};
  int lane = threadIdx.x & 31, half = lane >> 4;
  int mr = tm * 16 + (lane & 15);       // A-fragment row
  int nc = tn * 16 + (lane & 15);       // B/C-fragment column
  v8f acc = {};
#pragma unroll
  for (int k = 0; k < 4; ++k) {
    const float* T  = Ts[k];
    const float* Wk = W + k * 32 * 32;
    v16h a, b;
#pragma unroll
    for (int i = 0; i < 16; ++i) {
      int kk = kidx16(i, half);
      a[i] = (_Float16)T[mr * 32 + kk];
      b[i] = (_Float16)Wk[kk * 32 + nc];
    }
    acc = __builtin_amdgcn_wmma_f32_16x16x32_f16(false, a, false, b, (short)0, acc,
                                                 false, false);
  }
  float bv = bias[nc];
#pragma unroll
  for (int r = 0; r < 8; ++r)
    out[(tm * 16 + r + half * 8) * 32 + nc] = acc[r] + bv;   // C layout: M=r(+8 for half 1)
}

// Layer-2 dense (32 -> 1): per-node dot product over 4x32 = 128 terms.
__global__ void lag_dense_out1(const float* __restrict__ T0, const float* __restrict__ T1,
                               const float* __restrict__ T2, const float* __restrict__ T3,
                               const float* __restrict__ W, const float* __restrict__ bias,
                               float* __restrict__ out, int Nn) {
  int n = blockIdx.x * blockDim.x + threadIdx.x;
  if (n >= Nn) return;
  float s = bias[0];
#pragma unroll 8
  for (int c = 0; c < 32; ++c) {
    s += T0[n * 32 + c] * W[c]       + T1[n * 32 + c] * W[32 + c]
       + T2[n * 32 + c] * W[64 + c]  + T3[n * 32 + c] * W[96 + c];
  }
  out[n] = s;
}

// Graclus pair pooling: x1[b,e,c] = max(x[b,2e,c], x[b,2e+1,c]).
__global__ void pool_max(const float* __restrict__ in, float* __restrict__ out,
                         int N2n, int E2c, int Ec) {
  int t = blockIdx.x * blockDim.x + threadIdx.x;
  if (t >= N2n * 32) return;
  int n2 = t >> 5, c = t & 31;
  int b = n2 / E2c, e = n2 % E2c;
  int i0 = (b * Ec + 2 * e) * 32 + c;
  out[t] = fmaxf(in[i0], in[i0 + 32]);
}

// ---------------------------------------------------------------------------
// BatchNorm: two-pass (per-channel sum & sumsq via atomics, then apply+act).
// ---------------------------------------------------------------------------
__global__ void bn_zero(float* stats, int n) {
  int t = blockIdx.x * blockDim.x + threadIdx.x;
  if (t < n) stats[t] = 0.f;
}

// Requires gridDim.x*blockDim.x to be a multiple of C.
__global__ void bn_stats(const float* __restrict__ X, float* __restrict__ stats,
                         int Nrows, int C) {
  int t = blockIdx.x * blockDim.x + threadIdx.x;
  int total = gridDim.x * blockDim.x;
  int c = t % C;
  int r = t / C;
  int rs = total / C;
  float s = 0.f, s2 = 0.f;
  for (; r < Nrows; r += rs) {
    float v = X[r * C + c];
    s += v; s2 += v * v;
  }
  atomicAdd(&stats[c], s);
  atomicAdd(&stats[C + c], s2);
}

// y = g*(x-mean)*rsqrt(var+eps)+b, then LeakyReLU(slope) (slope=0 -> ReLU).
__global__ void bn_apply(float* __restrict__ X, const float* __restrict__ stats,
                         const float* __restrict__ g, const float* __restrict__ b,
                         int Nrows, int C, float slope) {
  int t = blockIdx.x * blockDim.x + threadIdx.x;
  if (t >= Nrows * C) return;
  int c = t % C;
  float invN = 1.f / (float)Nrows;
  float m = stats[c] * invN;
  float var = stats[C + c] * invN - m * m;
  float y = g[c] * (X[t] - m) * rsqrtf(var + BN_EPS) + b[c];
  X[t] = (y > 0.f) ? y : slope * y;
}

// ---------------------------------------------------------------------------
// Generic f32 GEMM via f16 WMMA: C[M,N] = A[M,K]@B[K,N] + bias (row-major).
// One wave per 16x16 output tile; K remainder zero-filled. M,N multiples of 16.
// ---------------------------------------------------------------------------
__global__ void gemm_bias_wmma(const float* __restrict__ A, const float* __restrict__ B,
                               const float* __restrict__ bias, float* __restrict__ C,
                               int M, int N, int K) {
  int wave = (blockIdx.x * blockDim.x + threadIdx.x) >> 5;
  int ntn = N >> 4;
  int tiles = (M >> 4) * ntn;
  if (wave >= tiles) return;
  int tm = wave / ntn, tn = wave % ntn;
  int lane = threadIdx.x & 31, half = lane >> 4;
  int mr = tm * 16 + (lane & 15);
  int nc = tn * 16 + (lane & 15);
  v8f acc = {};
  for (int k0 = 0; k0 < K; k0 += 32) {
    v16h a, b;
#pragma unroll
    for (int i = 0; i < 16; ++i) {
      int k = k0 + kidx16(i, half);
      a[i] = (k < K) ? (_Float16)A[mr * K + k] : (_Float16)0.0f;
      b[i] = (k < K) ? (_Float16)B[k * N + nc] : (_Float16)0.0f;
    }
    acc = __builtin_amdgcn_wmma_f32_16x16x32_f16(false, a, false, b, (short)0, acc,
                                                 false, false);
  }
  float bv = bias[nc];
#pragma unroll
  for (int r = 0; r < 8; ++r)
    C[(tm * 16 + r + half * 8) * N + nc] = acc[r] + bv;
}

// Final head: out[b] = h2[b,:] . lin3_W + lin3_b
__global__ void final_lin(const float* __restrict__ H, const float* __restrict__ W,
                          const float* __restrict__ bias, float* __restrict__ out, int Bc) {
  int b = blockIdx.x * blockDim.x + threadIdx.x;
  if (b >= Bc) return;
  float s = bias[0];
#pragma unroll 8
  for (int c = 0; c < 128; ++c) s += H[b * 128 + c] * W[c];
  out[b] = s;
}

// ---------------------------------------------------------------------------
extern "C" void kernel_launch(void* const* d_in, const int* in_sizes, int n_in,
                              void* d_out, int out_size, void* d_ws, size_t ws_size,
                              hipStream_t stream) {
  const int Bc = 64, Ec = 8978, E2c = 4489;
  const int N1 = Bc * Ec;        // 574592
  const int N2 = Bc * E2c;       // 287296

  // ---- inputs (setup_inputs order) ----
  const float* x_s   = (const float*)d_in[0];
  const int*   src1  = (const int*)  d_in[1];            // ei1[0,:]; dst[e]==e/16
  const float* ew1   = (const float*)d_in[2];
  const int*   src2  = (const int*)  d_in[3];
  const float* ew2   = (const float*)d_in[4];
  const float* W0    = (const float*)d_in[5];
  const float* b0    = (const float*)d_in[6];
  const float* g0    = (const float*)d_in[7];
  const float* be0   = (const float*)d_in[8];
  const float* W1    = (const float*)d_in[9];
  const float* b1    = (const float*)d_in[10];
  const float* g1    = (const float*)d_in[11];
  const float* be1   = (const float*)d_in[12];
  const float* W2    = (const float*)d_in[13];
  const float* b2    = (const float*)d_in[14];
  const float* g2    = (const float*)d_in[15];
  const float* be2   = (const float*)d_in[16];
  const float* lin1W = (const float*)d_in[17];
  const float* lin1b = (const float*)d_in[18];
  const float* bn1g  = (const float*)d_in[19];
  const float* bn1b  = (const float*)d_in[20];
  const float* lin2W = (const float*)d_in[21];
  const float* lin2b = (const float*)d_in[22];
  const float* bn2g  = (const float*)d_in[23];
  const float* bn2b  = (const float*)d_in[24];
  const float* lin3W = (const float*)d_in[25];
  const float* lin3b = (const float*)d_in[26];
  float* out = (float*)d_out;

  // ---- workspace layout (floats) ----
  float* ws = (float*)d_ws;
  float* out0 = ws;                                  // N1*32 (reused as layer1 out)
  float* x1   = out0 + (size_t)N1 * 32;              // N2*32 (reused as x_flat/out2)
  float* Tp   = x1   + (size_t)N2 * 32;              // N2*32
  float* Tc   = Tp   + (size_t)N2 * 32;              // N2*32
  float* Tn   = Tc   + (size_t)N2 * 32;              // N2*32
  float* stats = Tn  + (size_t)N2 * 32;              // 512
  float* h1    = stats + 1024;                       // 64*256
  float* h2    = h1 + 64 * 256;                      // 64*128

  const int TB = 256;
  const int statGrid = 512;   // 512*256 threads: multiple of every C used

  // ===================== Layer 0 (1 -> 32 ch, N1 nodes) =====================
  float* t1 = Tp;  float* t2 = Tc;  float* t3 = Tn;   // N1 <= N2*32, safe reuse
  lag_step_c1<<<(N1 + TB - 1) / TB, TB, 0, stream>>>(x_s, x_s, src1, ew1, t1, N1, 1.f, 0.f, 1.f);
  lag_step_c1<<<(N1 + TB - 1) / TB, TB, 0, stream>>>(t1, x_s, src1, ew1, t2, N1, 3.f, 1.f, 0.5f);
  lag_step_c1<<<(N1 + TB - 1) / TB, TB, 0, stream>>>(t2, t1, src1, ew1, t3, N1, 5.f, 2.f, 1.f / 3.f);
  lag_dense_in1<<<(N1 * 32) / TB, TB, 0, stream>>>(x_s, t1, t2, t3, W0, b0, out0, N1);
  bn_zero<<<1, 64, 0, stream>>>(stats, 64);
  bn_stats<<<statGrid, TB, 0, stream>>>(out0, stats, N1, 32);
  bn_apply<<<(N1 * 32) / TB, TB, 0, stream>>>(out0, stats, g0, be0, N1, 32, LEAKY_SLOPE);
  pool_max<<<(N2 * 32) / TB, TB, 0, stream>>>(out0, x1, N2, E2c, Ec);

  // ===================== Layer 1 (32 -> 32 ch, N2 nodes) ====================
  lag_step_c32<<<(N2 * 32) / TB, TB, 0, stream>>>(x1, x1, src2, ew2, Tp, N2, 1.f, 0.f, 1.f);
  lag_step_c32<<<(N2 * 32) / TB, TB, 0, stream>>>(Tp, x1, src2, ew2, Tc, N2, 3.f, 1.f, 0.5f);
  lag_step_c32<<<(N2 * 32) / TB, TB, 0, stream>>>(Tc, Tp, src2, ew2, Tn, N2, 5.f, 2.f, 1.f / 3.f);
  float* out1 = out0;  // reuse
  {
    int tiles = (N2 / 16) * 2;               // waves
    lag_dense32_wmma<<<(tiles + 7) / 8, TB, 0, stream>>>(x1, Tp, Tc, Tn, W1, b1, out1, N2);
  }
  bn_zero<<<1, 64, 0, stream>>>(stats, 64);
  bn_stats<<<statGrid, TB, 0, stream>>>(out1, stats, N2, 32);
  bn_apply<<<(N2 * 32) / TB, TB, 0, stream>>>(out1, stats, g1, be1, N2, 32, LEAKY_SLOPE);

  // ===================== Layer 2 (32 -> 1 ch, N2 nodes) =====================
  lag_step_c32<<<(N2 * 32) / TB, TB, 0, stream>>>(out1, out1, src2, ew2, Tp, N2, 1.f, 0.f, 1.f);
  lag_step_c32<<<(N2 * 32) / TB, TB, 0, stream>>>(Tp, out1, src2, ew2, Tc, N2, 3.f, 1.f, 0.5f);
  lag_step_c32<<<(N2 * 32) / TB, TB, 0, stream>>>(Tc, Tp, src2, ew2, Tn, N2, 5.f, 2.f, 1.f / 3.f);
  float* x_flat = x1;  // out2 == x reshaped [64, 4489]
  lag_dense_out1<<<(N2 + TB - 1) / TB, TB, 0, stream>>>(out1, Tp, Tc, Tn, W2, b2, x_flat, N2);
  bn_zero<<<1, 64, 0, stream>>>(stats, 64);
  bn_stats<<<statGrid, TB, 0, stream>>>(x_flat, stats, N2, 1);
  bn_apply<<<(N2 + TB - 1) / TB, TB, 0, stream>>>(x_flat, stats, g2, be2, N2, 1, LEAKY_SLOPE);

  // ===================== MLP head ==========================================
  // lin1: [64,4489]@[4489,256]
  {
    int tiles = (64 / 16) * (256 / 16);      // 64 waves
    gemm_bias_wmma<<<(tiles + 7) / 8, TB, 0, stream>>>(x_flat, lin1W, lin1b, h1, 64, 256, 4489);
  }
  bn_zero<<<2, 256, 0, stream>>>(stats, 512);
  bn_stats<<<64, TB, 0, stream>>>(h1, stats, 64, 256);
  bn_apply<<<(64 * 256) / TB, TB, 0, stream>>>(h1, stats, bn1g, bn1b, 64, 256, 0.f);
  // lin2: [64,256]@[256,128]
  {
    int tiles = (64 / 16) * (128 / 16);      // 32 waves
    gemm_bias_wmma<<<(tiles + 7) / 8, TB, 0, stream>>>(h1, lin2W, lin2b, h2, 64, 128, 256);
  }
  bn_zero<<<1, 256, 0, stream>>>(stats, 256);
  bn_stats<<<64, TB, 0, stream>>>(h2, stats, 64, 128);
  bn_apply<<<(64 * 128) / TB, TB, 0, stream>>>(h2, stats, bn2g, bn2b, 64, 128, 0.f);
  // lin3: [64,128]@[128,1]
  final_lin<<<1, 64, 0, stream>>>(h2, lin3W, lin3b, out, Bc);
}
